// BahdanauAttention_12506944766139
// MI455X (gfx1250) — compile-verified, matrix operand order checked by reading, not run on hardware
//
#include <hip/hip_runtime.h>
#include <hip/hip_bf16.h>
#include <math.h>

// Problem sizes (fixed by the reference)
#define B_SZ 32
#define S_SZ 2048
#define D_SZ 1024
#define KT_N (D_SZ / 32)   // 32 K-tiles of 32 for wmma bf16
#define NT_N (D_SZ / 16)   // 64 N-tiles of 16
#define APAD 24            // bf16 elements of row padding (48B) -> conflict-free LDS
#define AROW (D_SZ + APAD) // 1048 bf16 per LDS row
#define NSPLIT 16          // deterministic split-K for context reduction

typedef __attribute__((ext_vector_type(16))) __bf16 v16bf;
typedef __attribute__((ext_vector_type(8)))  __bf16 v8bf;
typedef __attribute__((ext_vector_type(8)))  float  v8f;
typedef __attribute__((ext_vector_type(4)))  float  v4f;
typedef __attribute__((ext_vector_type(4)))  unsigned int v4u;

union V16 { v16bf v; v8bf h[2]; };

__device__ __forceinline__ unsigned short f2bf_bits(float f) {
  unsigned int u = __builtin_bit_cast(unsigned int, f);
  u += 0x7FFFu + ((u >> 16) & 1u);           // round-to-nearest-even
  return (unsigned short)(u >> 16);
}

// ---------------------------------------------------------------------------
// K1: q[b,d] = dec_input[b]·W[:,d] + bW[d] + bU[d]   (bU folded into query)
// ---------------------------------------------------------------------------
__global__ void k_query(const float* __restrict__ x, const float* __restrict__ W,
                        const float* __restrict__ bW, const float* __restrict__ bU,
                        float* __restrict__ q) {
  __shared__ float sx[D_SZ];
  const int b = blockIdx.x, tid = threadIdx.x;
  for (int i = tid; i < D_SZ; i += 256) sx[i] = x[b * D_SZ + i];
  __syncthreads();
  const int d0 = tid * 4;
  v4f acc = {0.f, 0.f, 0.f, 0.f};
  for (int k = 0; k < D_SZ; ++k) {
    const float xk = sx[k];
    const v4f w = *(const v4f*)(W + (size_t)k * D_SZ + d0);
    acc += xk * w;
  }
  const v4f bw = *(const v4f*)(bW + d0);
  const v4f bu = *(const v4f*)(bU + d0);
  *(v4f*)(q + (size_t)b * D_SZ + d0) = acc + bw + bu;
}

// ---------------------------------------------------------------------------
// K2: pack U (fp32, row-major [K=1024][N=1024]) into bf16 B-fragments,
// pre-swizzled to the v_wmma_f32_16x16x32_bf16 lane layout:
//   lane holds column n = nt*16 + (lane&15); off = (lane<16)?0:8
//   uints p=0..3 : K = kt*32 + off + 2p, 2p+1
//   uints p=4..7 : K = kt*32 + 16 + off + 2(p-4), +1
// ---------------------------------------------------------------------------
__global__ void k_packU(const float* __restrict__ U, unsigned int* __restrict__ Ub) {
  const int tid  = blockIdx.x * blockDim.x + threadIdx.x;
  const int lane = tid & 31;
  const int tile = tid >> 5;               // nt*KT_N + kt, 0..2047
  const int kt   = tile & (KT_N - 1);
  const int nt   = tile >> 5;
  const int n    = nt * 16 + (lane & 15);
  const int off  = (lane < 16) ? 0 : 8;
  unsigned int out[8];
#pragma unroll
  for (int p = 0; p < 8; ++p) {
    const int kbase = (p < 4) ? (kt * 32 + off + 2 * p)
                              : (kt * 32 + 16 + off + 2 * (p - 4));
    const unsigned short lo = f2bf_bits(U[(size_t)(kbase + 0) * D_SZ + n]);
    const unsigned short hi = f2bf_bits(U[(size_t)(kbase + 1) * D_SZ + n]);
    out[p] = (unsigned int)lo | ((unsigned int)hi << 16);
  }
  unsigned int* dst = Ub + (size_t)tile * 256 + lane * 8;
  *(v4u*)(dst)     = (v4u){out[0], out[1], out[2], out[3]};
  *(v4u*)(dst + 4) = (v4u){out[4], out[5], out[6], out[7]};
}

// ---------------------------------------------------------------------------
// K3 (main): per 16-row (b, s-tile) slab:
//   values = enc_slab @ U  (bf16 WMMA, f32 accum)
//   scores[row] = sum_n tanh(q[b,n] + values[row,n]) * V[n]
// Fused: the [B,S,D] values/energy tensors are never materialized.
// ---------------------------------------------------------------------------
__global__ void __launch_bounds__(256, 2)
k_energy_scores(const float* __restrict__ enc, const unsigned int* __restrict__ Ub,
                const float* __restrict__ q, const float* __restrict__ V,
                float* __restrict__ scores) {
  __shared__ __bf16 sA[16 * AROW];
  __shared__ float  sSc[16];

  const int b   = blockIdx.y;
  const int st  = blockIdx.x;
  const int tid = threadIdx.x;
  const int s0  = st * 16;

  // Stage A-slab: 16 rows x 1024 fp32 -> bf16 in padded LDS
  const float* src = enc + ((size_t)b * S_SZ + s0) * D_SZ;
  for (int idx = tid; idx < 16 * 256; idx += 256) {
    const int r = idx >> 8;      // row 0..15
    const int c = idx & 255;     // float4 index in row
    const v4f f = *(const v4f*)(src + (size_t)r * D_SZ + c * 4);
    const unsigned int lo = (unsigned int)f2bf_bits(f.x) | ((unsigned int)f2bf_bits(f.y) << 16);
    const unsigned int hi = (unsigned int)f2bf_bits(f.z) | ((unsigned int)f2bf_bits(f.w) << 16);
    const unsigned long long pk = (unsigned long long)lo | ((unsigned long long)hi << 32);
    *(unsigned long long*)((char*)sA + ((size_t)r * AROW + c * 4) * 2) = pk;
  }
  if (tid < 16) sSc[tid] = 0.f;
  __syncthreads();

  const int lane = tid & 31;
  const int wave = tid >> 5;
  const int mrow = lane & 15;
  const int off  = (lane < 16) ? 0 : 8;
  const __bf16* arow = &sA[(size_t)mrow * AROW];

  float psum[8] = {0.f, 0.f, 0.f, 0.f, 0.f, 0.f, 0.f, 0.f};

  for (int ni = 0; ni < 8; ++ni) {
    const int nt = wave + ni * 8;          // 8 waves cover 64 N-tiles
    const int n  = nt * 16 + mrow;
    v8f acc = {0.f, 0.f, 0.f, 0.f, 0.f, 0.f, 0.f, 0.f};
    const unsigned int* bptr = Ub + (size_t)(nt * KT_N) * 256 + lane * 8;
    for (int kt = 0; kt < KT_N; ++kt) {
      V16 a, bb;
      a.h[0]  = *(const v8bf*)(arow + kt * 32 + off);        // K = off..off+7
      a.h[1]  = *(const v8bf*)(arow + kt * 32 + 16 + off);   // K = 16+off..+7
      bb.h[0] = *(const v8bf*)(bptr);
      bb.h[1] = *(const v8bf*)(bptr + 4);
      bptr += 256;
      acc = __builtin_amdgcn_wmma_f32_16x16x32_bf16(
          false, a.v, false, bb.v, (short)0, acc, false, false);
    }
    const float qn = q[(size_t)b * D_SZ + n];   // query + bU, precombined
    const float vn = V[n];
#pragma unroll
    for (int i = 0; i < 8; ++i)
      psum[i] += tanhf(acc[i] + qn) * vn;
  }

  // Reduce each row's partial across its 16-lane group, then across waves.
#pragma unroll
  for (int i = 0; i < 8; ++i) {
    float r = psum[i];
    r += __shfl_xor(r, 1, 32);
    r += __shfl_xor(r, 2, 32);
    r += __shfl_xor(r, 4, 32);
    r += __shfl_xor(r, 8, 32);
    if (lane == 0)  atomicAdd(&sSc[i], r);       // rows 0..7
    if (lane == 16) atomicAdd(&sSc[i + 8], r);   // rows 8..15
  }
  __syncthreads();
  if (tid < 16) scores[(size_t)b * S_SZ + s0 + tid] = sSc[tid];
}

// ---------------------------------------------------------------------------
// K4: softmax over S per batch.  (bV dropped: softmax is shift-invariant.)
// ---------------------------------------------------------------------------
__global__ void k_softmax(const float* __restrict__ scores, float* __restrict__ attn) {
  __shared__ float red[8];
  const int b = blockIdx.x, tid = threadIdx.x;
  float v[8];
  float mx = -3.0e38f;
#pragma unroll
  for (int j = 0; j < 8; ++j) {
    v[j] = scores[(size_t)b * S_SZ + tid + j * 256];
    mx = fmaxf(mx, v[j]);
  }
  for (int o = 16; o; o >>= 1) mx = fmaxf(mx, __shfl_xor(mx, o, 32));
  if ((tid & 31) == 0) red[tid >> 5] = mx;
  __syncthreads();
  float m = red[0];
#pragma unroll
  for (int i = 1; i < 8; ++i) m = fmaxf(m, red[i]);
  __syncthreads();

  float s = 0.f;
#pragma unroll
  for (int j = 0; j < 8; ++j) { v[j] = __expf(v[j] - m); s += v[j]; }
  for (int o = 16; o; o >>= 1) s += __shfl_xor(s, o, 32);
  if ((tid & 31) == 0) red[tid >> 5] = s;
  __syncthreads();
  float tot = 0.f;
#pragma unroll
  for (int i = 0; i < 8; ++i) tot += red[i];
  const float inv = 1.f / tot;
#pragma unroll
  for (int j = 0; j < 8; ++j)
    attn[(size_t)b * S_SZ + tid + j * 256] = v[j] * inv;
}

// ---------------------------------------------------------------------------
// K5: context split-K partials: part[b,sp,:] = sum_{s in split} attn * enc row
// Deterministic (no fp32 global atomics).
// ---------------------------------------------------------------------------
__global__ void k_ctx_part(const float* __restrict__ enc, const float* __restrict__ attn,
                           float* __restrict__ part) {
  __shared__ float sa[S_SZ / NSPLIT];   // 128 weights
  const int b = blockIdx.x, sp = blockIdx.y, tid = threadIdx.x;
  const int sbeg = sp * (S_SZ / NSPLIT);
  if (tid < S_SZ / NSPLIT) sa[tid] = attn[(size_t)b * S_SZ + sbeg + tid];
  __syncthreads();
  v4f acc = {0.f, 0.f, 0.f, 0.f};
  const float* base = enc + ((size_t)b * S_SZ + sbeg) * D_SZ + tid * 4;
  for (int s = 0; s < S_SZ / NSPLIT; ++s) {
    const v4f e = *(const v4f*)(base + (size_t)s * D_SZ);
    acc += sa[s] * e;
  }
  *(v4f*)(part + ((size_t)(b * NSPLIT + sp)) * D_SZ + tid * 4) = acc;
}

// K6: sum the NSPLIT partials into the context output.
__global__ void k_ctx_reduce(const float* __restrict__ part, float* __restrict__ ctx) {
  const int b = blockIdx.x, tid = threadIdx.x;
  v4f acc = {0.f, 0.f, 0.f, 0.f};
#pragma unroll
  for (int sp = 0; sp < NSPLIT; ++sp)
    acc += *(const v4f*)(part + ((size_t)(b * NSPLIT + sp)) * D_SZ + tid * 4);
  *(v4f*)(ctx + (size_t)b * D_SZ + tid * 4) = acc;
}

// ---------------------------------------------------------------------------
extern "C" void kernel_launch(void* const* d_in, const int* in_sizes, int n_in,
                              void* d_out, int out_size, void* d_ws, size_t ws_size,
                              hipStream_t stream) {
  const float* dec = (const float*)d_in[0];  // [32,1,1024]
  const float* enc = (const float*)d_in[1];  // [32,2048,1024]
  const float* W   = (const float*)d_in[2];  // [1024,1024]
  const float* bW  = (const float*)d_in[3];  // [1024]
  const float* U   = (const float*)d_in[4];  // [1024,1024]
  const float* bU  = (const float*)d_in[5];  // [1024]
  const float* V   = (const float*)d_in[6];  // [1024,1]
  // d_in[7] = bV : mathematically irrelevant (softmax shift-invariance)

  char* ws = (char*)d_ws;
  float*        qbuf   = (float*)ws;                  // 128 KB  (query + bU)
  float*        scores = (float*)(ws + 131072);       // 256 KB
  unsigned int* Ub     = (unsigned int*)(ws + 393216);// 2 MB  (packed bf16 U)
  float*        part   = (float*)(ws + 2490368);      // 2 MB  (context split-K)

  float* ctx  = (float*)d_out;                 // [32,1,1024]
  float* attn = (float*)d_out + B_SZ * D_SZ;   // [32,2048]

  k_query<<<B_SZ, 256, 0, stream>>>(dec, W, bW, bU, qbuf);
  k_packU<<<256, 256, 0, stream>>>(U, Ub);
  k_energy_scores<<<dim3(S_SZ / 16, B_SZ), 256, 0, stream>>>(enc, Ub, qbuf, V, scores);
  k_softmax<<<B_SZ, 256, 0, stream>>>(scores, attn);
  k_ctx_part<<<dim3(B_SZ, NSPLIT), 256, 0, stream>>>(enc, attn, part);
  k_ctx_reduce<<<B_SZ, 256, 0, stream>>>(part, ctx);
}
